// TextDecoder_30906584662577
// MI455X (gfx1250) — matrix operand order repaired and checked by reading, not compiled, not run on hardware
//
#include <hip/hip_runtime.h>
#include <hip/hip_bf16.h>

// ---------------------------------------------------------------------------
// Problem dims (compile-time constants from the reference)
// ---------------------------------------------------------------------------
#define B_   32
#define T_   128
#define H_   1024
#define E_   256
#define V_   32000
#define G4H  4096   // 4*H

typedef __attribute__((ext_vector_type(16))) __bf16 v16bf;
typedef __attribute__((ext_vector_type(8)))  __bf16 v8bf;
typedef __attribute__((ext_vector_type(8)))  float  v8f;
typedef __attribute__((ext_vector_type(4)))  unsigned int v4u;

// ---------------------------------------------------------------------------
// WMMA fragment loaders (wave32, per CDNA5 ISA 7.12.2)
//
// A (16x32 bf16, M x K), row-major A[m][k]:
//   lane m (0..15):  VGPR0-3 = K k0..k0+7,   VGPR4-7 = K k0+16..k0+23
//   lane m+16:       VGPR0-3 = K k0+8..+15,  VGPR4-7 = K k0+24..+31
//   -> two contiguous b128 loads per lane.
//
// B (32x16 bf16, K x N) with B[k][n] = W[n][k], W row-major [N][K]:
//   lane n (0..15):  VGPR0-3 = K k0..k0+7,   VGPR4-7 = K k0+16..k0+23 of column n
//   lane n+16:       VGPR0-3 = K k0+8..+15,  VGPR4-7 = K k0+24..+31
//   -> W row-major IS the needed layout; two contiguous b128 loads per lane.
// These work identically on global or LDS panels (addrspace inferred after
// inlining -> global_load_b128 or ds_load_b128).
// ---------------------------------------------------------------------------
__device__ __forceinline__ v16bf load_a_frag(const __bf16* __restrict__ A,
                                             size_t lda, int m0, int k0, int lane) {
  const int m    = m0 + (lane & 15);
  const int koff = (lane >> 4) ? 8 : 0;
  const __bf16* p = A + (size_t)m * lda + (size_t)(k0 + koff);
  v8bf lo = *(const v8bf*)(p);
  v8bf hi = *(const v8bf*)(p + 16);
  return __builtin_shufflevector(lo, hi, 0,1,2,3,4,5,6,7,8,9,10,11,12,13,14,15);
}

__device__ __forceinline__ v16bf load_b_frag(const __bf16* __restrict__ W,
                                             size_t ldw, int n0, int k0, int lane) {
  const int n    = n0 + (lane & 15);
  const int koff = (lane >> 4) ? 8 : 0;
  const __bf16* p = W + (size_t)n * ldw + (size_t)(k0 + koff);
  v8bf lo = *(const v8bf*)(p);
  v8bf hi = *(const v8bf*)(p + 16);
  return __builtin_shufflevector(lo, hi, 0,1,2,3,4,5,6,7,8,9,10,11,12,13,14,15);
}

__device__ __forceinline__ v8f wmma_bf16(v16bf a, v16bf b, v8f c) {
  return __builtin_amdgcn_wmma_f32_16x16x32_bf16(
      /*neg_a=*/false, a, /*neg_b=*/false, b,
      /*c_mod=*/(short)0, c, /*reuse_a=*/false, /*reuse_b=*/false);
}

// ---------------------------------------------------------------------------
// Prep kernels
// ---------------------------------------------------------------------------
__global__ void f32_to_bf16_kernel(const float* __restrict__ src,
                                   __bf16* __restrict__ dst, size_t n) {
  size_t i      = (size_t)blockIdx.x * blockDim.x + threadIdx.x;
  size_t stride = (size_t)gridDim.x * blockDim.x;
  for (; i < n; i += stride) dst[i] = (__bf16)src[i];
}

// emb_bf[(b*T+t)*E + e] = bf16(emb_table[targets[b*T+t]*E + e])
__global__ void gather_emb_kernel(const int* __restrict__ targets,
                                  const float* __restrict__ table,
                                  __bf16* __restrict__ dst) {
  size_t i = (size_t)blockIdx.x * blockDim.x + threadIdx.x;
  if (i >= (size_t)B_ * T_ * E_) return;
  size_t row = i >> 8;          // / E_
  int    e   = (int)(i & (E_ - 1));
  int    tok = targets[row];
  dst[i] = (__bf16)table[(size_t)tok * E_ + e];
}

// gbias = b_ih + b_hh ; h0 = bf16(context) ; c0 = 0
__global__ void init_state_kernel(const float* __restrict__ b_ih,
                                  const float* __restrict__ b_hh,
                                  float* __restrict__ gbias,
                                  const float* __restrict__ context,
                                  __bf16* __restrict__ h_bf,
                                  float* __restrict__ c) {
  int i = blockIdx.x * blockDim.x + threadIdx.x;
  if (i < G4H) gbias[i] = b_ih[i] + b_hh[i];
  if (i < B_ * H_) {
    h_bf[i] = (__bf16)context[i];
    c[i]    = 0.0f;
  }
}

// ---------------------------------------------------------------------------
// LSTM step GEMM (serial-critical, latency-optimized: one 16x16 tile per wave,
// direct global b128 loads pipelined against WMMA -- verified good asm).
//   g[b][n] = h @ W_hh^T  +  emb_t @ W_ih^T  +  (b_ih + b_hh)
// M=32 (2 tiles) x N=4096 (256 tiles) -> 512 waves = 64 blocks x 8 waves.
// ---------------------------------------------------------------------------
__global__ __launch_bounds__(256)
void lstm_gemm_kernel(const __bf16* __restrict__ h_bf,    // [B, H]
                      const __bf16* __restrict__ Whh,     // [4H, H] row-major
                      const __bf16* __restrict__ emb_bf,  // [B*T, E]
                      const __bf16* __restrict__ Wih,     // [4H, E] row-major
                      const float*  __restrict__ gbias,   // [4H]
                      float*        __restrict__ g,       // [B, 4H]
                      int t) {
  const int lane = threadIdx.x & 31;
  const int wid  = blockIdx.x * (blockDim.x >> 5) + (threadIdx.x >> 5);
  const int m0   = (wid & 1) * 16;
  const int n0   = (wid >> 1) * 16;

  v8f acc = {};

  // hidden-to-gate contribution: K = H
  for (int k0 = 0; k0 < H_; k0 += 32) {
    v16bf a = load_a_frag(h_bf, H_, m0, k0, lane);
    v16bf b = load_b_frag(Whh, H_, n0, k0, lane);
    acc = wmma_bf16(a, b, acc);
  }
  // input-to-gate contribution (fused, avoids materializing xg): K = E
  // A row for batch m is emb_bf[(m*T + t)*E + k] -> base = emb_bf + t*E, lda = T*E
  const __bf16* Aemb = emb_bf + (size_t)t * E_;
  for (int k0 = 0; k0 < E_; k0 += 32) {
    v16bf a = load_a_frag(Aemb, (size_t)T_ * E_, m0, k0, lane);
    v16bf b = load_b_frag(Wih, E_, n0, k0, lane);
    acc = wmma_bf16(a, b, acc);
  }

  const int moff = (lane >> 4) ? 8 : 0;
  const int col  = n0 + (lane & 15);
  const float bv = gbias[col];
#pragma unroll
  for (int r = 0; r < 8; ++r) {
    const int m = m0 + moff + r;
    g[(size_t)m * G4H + col] = acc[r] + bv;
  }
}

// ---------------------------------------------------------------------------
// LSTM gate nonlinearity + state update (torch gate order i,f,g,o)
// ---------------------------------------------------------------------------
__global__ void lstm_gates_kernel(const float* __restrict__ g,   // [B, 4H]
                                  float*  __restrict__ c,        // [B, H]
                                  __bf16* __restrict__ h_bf,     // [B, H]
                                  __bf16* __restrict__ hs_bf,    // [B, T, H]
                                  int t) {
  int idx = blockIdx.x * blockDim.x + threadIdx.x;
  if (idx >= B_ * H_) return;
  const int b = idx >> 10;          // / H_
  const int j = idx & (H_ - 1);
  const float* gb = g + (size_t)b * G4H;
  const float iv = gb[j];
  const float fv = gb[H_ + j];
  const float gv = gb[2 * H_ + j];
  const float ov = gb[3 * H_ + j];

  const float si = 1.0f / (1.0f + __expf(-iv));
  const float sf = 1.0f / (1.0f + __expf(-fv));
  const float so = 1.0f / (1.0f + __expf(-ov));

  const float cn = sf * c[idx] + si * tanhf(gv);
  const float hn = so * tanhf(cn);

  c[idx] = cn;
  const __bf16 hb = (__bf16)hn;
  h_bf[idx] = hb;
  hs_bf[((size_t)b * T_ + t) * H_ + j] = hb;
}

// ---------------------------------------------------------------------------
// FC logits GEMM (throughput phase, 268 GFLOP): LDS-staged, double-buffered.
//
// Block = 8 waves computes a 128x256 macro-tile; per BK=32 panel the block
// stages A(128x32) + B(256x32) = 24 KB into LDS and executes 8x16 WMMAs
// -> ~85 flops per L2 byte (~3.2 GB total L2 traffic vs 8.4 GB unstaged).
// LDS: 2 buffers x (128+256) rows x 40-elt padded pitch x 2B = 60 KB
// (pitch 40 = 20 dwords -> the 16 rows of a b128 fragment read map to 16
//  distinct banks). Next panel is prefetched into registers so global loads
// overlap WMMA; one barrier per panel.
// ---------------------------------------------------------------------------
#define FC_BM  128
#define FC_BN  256
#define FC_BK  32
#define FC_LDP 40    // padded LDS row pitch in elements

__global__ __launch_bounds__(256)
void fc_gemm_kernel(const __bf16* __restrict__ hs_bf,  // [B*T, H]
                    const __bf16* __restrict__ Wfc,    // [V, H] row-major
                    const float*  __restrict__ fc_b,   // [V]
                    float*        __restrict__ out) {  // [B*T, V]
  __shared__ __bf16 As[2][FC_BM * FC_LDP];   // 2 x 10.0 KB
  __shared__ __bf16 Bs[2][FC_BN * FC_LDP];   // 2 x 20.0 KB

  const int tid  = threadIdx.x;
  const int lane = tid & 31;
  const int w    = tid >> 5;

  const int nb = blockIdx.x % (V_ / FC_BN);        // 125
  const int mb = blockIdx.x / (V_ / FC_BN);        // 32
  const size_t gm0 = (size_t)mb * FC_BM;
  const size_t gn0 = (size_t)nb * FC_BN;

  const int wm = (w & 1) * 64;    // wave tile row origin in block
  const int wn = (w >> 1) * 64;   // wave tile col origin in block

  v8f acc[4][4] = {};

  // ---- stage panel 0 ----
  {
    v4u ar[2], br[4];
#pragma unroll
    for (int i = 0; i < 2; ++i) {
      const int c = i * 256 + tid, row = c >> 2, q = c & 3;
      ar[i] = *(const v4u*)(hs_bf + (gm0 + row) * H_ + q * 8);
    }
#pragma unroll
    for (int i = 0; i < 4; ++i) {
      const int c = i * 256 + tid, row = c >> 2, q = c & 3;
      br[i] = *(const v4u*)(Wfc + (gn0 + row) * H_ + q * 8);
    }
#pragma unroll
    for (int i = 0; i < 2; ++i) {
      const int c = i * 256 + tid, row = c >> 2, q = c & 3;
      *(v4u*)(&As[0][row * FC_LDP + q * 8]) = ar[i];
    }
#pragma unroll
    for (int i = 0; i < 4; ++i) {
      const int c = i * 256 + tid, row = c >> 2, q = c & 3;
      *(v4u*)(&Bs[0][row * FC_LDP + q * 8]) = br[i];
    }
  }
  __syncthreads();

  int p = 0;
  const int NK = H_ / FC_BK;   // 32 panels
  for (int kt = 0; kt < NK; ++kt) {
    // Prefetch next panel into registers (loads in flight during WMMA).
    v4u ar[2], br[4];
    const bool more = (kt + 1 < NK);
    if (more) {
      const int k0 = (kt + 1) * FC_BK;
#pragma unroll
      for (int i = 0; i < 2; ++i) {
        const int c = i * 256 + tid, row = c >> 2, q = c & 3;
        ar[i] = *(const v4u*)(hs_bf + (gm0 + row) * H_ + k0 + q * 8);
      }
#pragma unroll
      for (int i = 0; i < 4; ++i) {
        const int c = i * 256 + tid, row = c >> 2, q = c & 3;
        br[i] = *(const v4u*)(Wfc + (gn0 + row) * H_ + k0 + q * 8);
      }
    }

    // Compute from current LDS buffer: 4 A frags x 4 B frags -> 16 WMMAs.
    {
      const __bf16* Ap = &As[p][0];
      const __bf16* Bp = &Bs[p][0];
      v16bf a[4];
#pragma unroll
      for (int i = 0; i < 4; ++i)
        a[i] = load_a_frag(Ap, FC_LDP, wm + 16 * i, 0, lane);
#pragma unroll
      for (int j = 0; j < 4; ++j) {
        v16bf b = load_b_frag(Bp, FC_LDP, wn + 16 * j, 0, lane);
#pragma unroll
        for (int i = 0; i < 4; ++i)
          acc[i][j] = wmma_bf16(a[i], b, acc[i][j]);
      }
    }

    // Commit prefetched panel into the other buffer, then one barrier.
    if (more) {
#pragma unroll
      for (int i = 0; i < 2; ++i) {
        const int c = i * 256 + tid, row = c >> 2, q = c & 3;
        *(v4u*)(&As[p ^ 1][row * FC_LDP + q * 8]) = ar[i];
      }
#pragma unroll
      for (int i = 0; i < 4; ++i) {
        const int c = i * 256 + tid, row = c >> 2, q = c & 3;
        *(v4u*)(&Bs[p ^ 1][row * FC_LDP + q * 8]) = br[i];
      }
    }
    __syncthreads();
    p ^= 1;
  }

  // Epilogue: D layout is lane l, vgpr r -> row r + 8*(l>=16), col l&15.
  const int moff = (lane >> 4) ? 8 : 0;
  const int cl   = lane & 15;
#pragma unroll
  for (int j = 0; j < 4; ++j) {
    const size_t col = gn0 + wn + 16 * j + cl;
    const float  bv  = fc_b[col];
#pragma unroll
    for (int i = 0; i < 4; ++i) {
#pragma unroll
      for (int r = 0; r < 8; ++r) {
        const size_t row = gm0 + wm + 16 * i + moff + r;
        out[row * V_ + col] = acc[i][j][r] + bv;
      }
    }
  }
}

// ---------------------------------------------------------------------------
// Host launcher
// ---------------------------------------------------------------------------
extern "C" void kernel_launch(void* const* d_in, const int* in_sizes, int n_in,
                              void* d_out, int out_size, void* d_ws, size_t ws_size,
                              hipStream_t stream) {
  (void)in_sizes; (void)n_in; (void)out_size; (void)ws_size;

  const float* context   = (const float*)d_in[0];   // [B, H]
  const int*   targets   = (const int*)  d_in[1];   // [B, T]
  const float* emb_table = (const float*)d_in[2];   // [V, E]
  const float* W_ih      = (const float*)d_in[3];   // [4H, E]
  const float* W_hh      = (const float*)d_in[4];   // [4H, H]
  const float* b_ih      = (const float*)d_in[5];   // [4H]
  const float* b_hh      = (const float*)d_in[6];   // [4H]
  const float* fc_W      = (const float*)d_in[7];   // [V, H]
  const float* fc_b      = (const float*)d_in[8];   // [V]
  float*       out       = (float*)d_out;           // [B, T, V]

  // Workspace bump allocator (256B aligned). Total ~87 MB.
  char*  ws  = (char*)d_ws;
  size_t off = 0;
  auto alloc = [&](size_t bytes) -> void* {
    off = (off + 255) & ~(size_t)255;
    void* p = ws + off;
    off += bytes;
    return p;
  };

  __bf16* Wih_bf = (__bf16*)alloc((size_t)G4H * E_ * 2);      // 2 MB
  __bf16* Whh_bf = (__bf16*)alloc((size_t)G4H * H_ * 2);      // 8 MB
  __bf16* Wfc_bf = (__bf16*)alloc((size_t)V_ * H_ * 2);       // 64 MB (L2-resident)
  __bf16* emb_bf = (__bf16*)alloc((size_t)B_ * T_ * E_ * 2);  // 2 MB
  float*  gbias  = (float*) alloc((size_t)G4H * 4);
  __bf16* h_bf   = (__bf16*)alloc((size_t)B_ * H_ * 2);
  float*  c_st   = (float*) alloc((size_t)B_ * H_ * 4);
  float*  g_buf  = (float*) alloc((size_t)B_ * G4H * 4);
  __bf16* hs_bf  = (__bf16*)alloc((size_t)B_ * T_ * H_ * 2);  // 8 MB

  // --- Prep: weight downconvert (inside the graph; deterministic) ---
  f32_to_bf16_kernel<<<2048, 256, 0, stream>>>(W_ih, Wih_bf, (size_t)G4H * E_);
  f32_to_bf16_kernel<<<4096, 256, 0, stream>>>(W_hh, Whh_bf, (size_t)G4H * H_);
  f32_to_bf16_kernel<<<8192, 256, 0, stream>>>(fc_W, Wfc_bf, (size_t)V_ * H_);
  gather_emb_kernel<<<(B_ * T_ * E_ + 255) / 256, 256, 0, stream>>>(
      targets, emb_table, emb_bf);
  init_state_kernel<<<(B_ * H_ + 255) / 256, 256, 0, stream>>>(
      b_ih, b_hh, gbias, context, h_bf, c_st);

  // --- Sequential LSTM scan: 128 steps, 2 launches each (stream-ordered) ---
  for (int t = 0; t < T_; ++t) {
    lstm_gemm_kernel<<<64, 256, 0, stream>>>(h_bf, Whh_bf, emb_bf, Wih_bf,
                                             gbias, g_buf, t);
    lstm_gates_kernel<<<(B_ * H_ + 255) / 256, 256, 0, stream>>>(
        g_buf, c_st, h_bf, hs_bf, t);
  }

  // --- Vocab projection: the dominant 268 GFLOP GEMM ---
  fc_gemm_kernel<<<4000, 256, 0, stream>>>(hs_bf, Wfc_bf, fc_b, out);
}
